// CheMeleonEncoder_9036611190784
// MI455X (gfx1250) — matrix operand order repaired and checked by reading, not compiled, not run on hardware
//
#include <hip/hip_runtime.h>

// ---------------------------------------------------------------------------
// Types for CDNA5 WMMA
// ---------------------------------------------------------------------------
typedef __attribute__((ext_vector_type(16))) __bf16 v16bf;
typedef __attribute__((ext_vector_type(8)))  __bf16 v8bf;
typedef __attribute__((ext_vector_type(8)))  float  v8f;

union FragBF { v16bf f; v8bf h[2]; };

// Native bf16 <-> f32 conversion (v_cvt_pk_bf16_f32 on gfx1250).
static __device__ __forceinline__ unsigned short f2bf(float f) {
    __bf16 b = (__bf16)f;
    return __builtin_bit_cast(unsigned short, b);
}
static __device__ __forceinline__ float bf2f(unsigned short h) {
    return (float)__builtin_bit_cast(__bf16, h);
}

// CDNA5 async global->LDS copy (ASYNCcnt tracked), per CDNA5 ISA.
// Each lane copies 16 bytes from its global address to its LDS address.
static __device__ __forceinline__ void async_copy_b128(const void* gptr, void* lptr) {
    unsigned lds = (unsigned)(unsigned long long)lptr;  // LDS byte offset (addr[31:0])
    asm volatile("global_load_async_to_lds_b128 %0, %1, off"
                 :: "v"(lds), "v"(gptr) : "memory");
}
// Async loads complete in order -> waiting for cnt<=N leaves the newest N
// copies in flight (double-buffer pipelining).
static __device__ __forceinline__ void wait_async_le0() {
    asm volatile("s_wait_asynccnt 0x0" ::: "memory");
}
static __device__ __forceinline__ void wait_async_le2() {
    asm volatile("s_wait_asynccnt 0x2" ::: "memory");
}
static __device__ __forceinline__ void wait_async_le4() {
    asm volatile("s_wait_asynccnt 0x4" ::: "memory");
}

// Problem constants
#define NATOMS 16384
#define NEDGES 32768
#define NMOLS  512
#define DV 72
#define DE 14
#define DH 2048
#define K_INIT  96    // 72 + 14 padded to 96
#define K_FINAL 2144  // 72 + 2048 padded to 2144

// ---------------------------------------------------------------------------
// Weight packing: f32 [Kin, N] -> bf16 TRANSPOSED [N, Kpad], zero padded K.
// Transposed layout makes the GEMM's B tile a contiguous row-major copy
// (async-LDS friendly, no in-LDS transpose needed).
// ---------------------------------------------------------------------------
__global__ void pack_wt_kernel(const float* __restrict__ W,
                               unsigned short* __restrict__ out,
                               int Kin, int Kpad, int N) {
    long long i = (long long)blockIdx.x * blockDim.x + threadIdx.x;
    long long total = (long long)N * Kpad;
    if (i >= total) return;
    int n = (int)(i / Kpad);
    int k = (int)(i - (long long)n * Kpad);
    out[i] = (k < Kin) ? f2bf(W[(long long)k * N + n]) : (unsigned short)0;
}

// ---------------------------------------------------------------------------
// WMMA GEMM: OUT[M,N] = act( bias + A @ B )
//   A_MODE 0: A = bf16 [M, K] row-major (plain, async-staged)
//   A_MODE 1: A[e,k] = k<72 ? V[edge_src[e],k] : (k<86 ? E[e,k-72] : 0)
//   A_MODE 2: A[r,k] = k<72 ? V[r,k] : (k<2120 ? Mv[r,k-72] : 0)
//   BIAS_MODE 0: none; 1: bf16 H0[M,N] tile; 2: f32 vector bias[N]
//   OUT_F32 0: bf16 out; 1: f32 out
// B is bf16 [N, K] (pre-transposed). Block: 256 threads (8 waves), tile
// 128x128, wave tile 64x32 = 4x2 v_wmma_f32_16x16x32_bf16, K step 32.
// Double-buffered LDS; async copies for tile k+1 stay in flight across the
// compute of tile k (s_wait_asynccnt <= per-tile-op-count).
// ---------------------------------------------------------------------------
template <int A_MODE, int BIAS_MODE, int OUT_F32>
__global__ __launch_bounds__(256)
void gemm_wmma_kernel(const unsigned short* __restrict__ Abf,
                      const float* __restrict__ Vp,
                      const float* __restrict__ Ep,
                      const int*   __restrict__ edge_src,
                      const float* __restrict__ Mvp,
                      const unsigned short* __restrict__ Bt,   // bf16 [N, K]
                      const unsigned short* __restrict__ H0b,  // bf16 [M, N]
                      const float* __restrict__ bias_vec,      // f32 [N]
                      void* __restrict__ Out,
                      int N, int K) {
    __shared__ __align__(16) unsigned short sA[2][128 * 32];  // [buf][row][k]
    __shared__ __align__(16) unsigned short sB[2][128 * 32];  // [buf][n][k]

    const int tid  = threadIdx.x;
    const int wave = tid >> 5;
    const int lane = tid & 31;
    const int halfsel = lane >> 4;    // 0: lanes 0-15, 1: lanes 16-31
    const int lrow    = lane & 15;

    const int row0 = blockIdx.y * 128;
    const int col0 = blockIdx.x * 128;
    const int wm = (wave & 1) * 64;   // wave row offset inside block tile
    const int wn = (wave >> 1) * 32;  // wave col offset inside block tile

    const int idx = tid * 16;         // 256 thr * 16 elems = 4096 = one tile
    const int r   = idx >> 5;         // tile row (same for all 16 elems)
    const int kk0 = idx & 31;         // 0 or 16

    // Stage tile kt into LDS buffers dA/dB (B always async; A async in mode 0)
    auto stage = [&](int kt, unsigned short* dA, unsigned short* dB) {
        const int kbase = kt << 5;
        {
            const unsigned short* g = Bt + (size_t)(col0 + r) * K + kbase + kk0;
            async_copy_b128(g,     dB + idx);
            async_copy_b128(g + 8, dB + idx + 8);
        }
        if (A_MODE == 0) {
            const unsigned short* g = Abf + (size_t)(row0 + r) * K + kbase + kk0;
            async_copy_b128(g,     dA + idx);
            async_copy_b128(g + 8, dA + idx + 8);
        } else {
            const int grow = row0 + r;
            int vrow = grow;
            if (A_MODE == 1) vrow = edge_src[grow];
#pragma unroll
            for (int i = 0; i < 16; ++i) {
                const int gk = kbase + kk0 + i;
                float val = 0.f;
                if (A_MODE == 1) {
                    if (gk < DV)           val = Vp[(size_t)vrow * DV + gk];
                    else if (gk < DV + DE) val = Ep[(size_t)grow * DE + (gk - DV)];
                } else {
                    if (gk < DV)           val = Vp[(size_t)grow * DV + gk];
                    else if (gk < DV + DH) val = Mvp[(size_t)grow * DH + (gk - DV)];
                }
                dA[idx + i] = f2bf(val);
            }
        }
    };

    v8f acc[4][2] = {};

    const int ksteps = K >> 5;
    stage(0, sA[0], sB[0]);                       // prologue: tile 0 in flight

    for (int kt = 0; kt < ksteps; ++kt) {
        const int cur = kt & 1;
        if (kt + 1 < ksteps) {
            stage(kt + 1, sA[1 - cur], sB[1 - cur]);  // next tile in flight
            if (A_MODE == 0) wait_async_le4();        // tile kt landed
            else             wait_async_le2();
        } else {
            wait_async_le0();
        }
        __syncthreads();                          // tile kt visible to all waves

        const unsigned short* cA = sA[cur];
        const unsigned short* cB = sB[cur];

        // ---- B fragments (2 N-tiles): lanes 0-15 K=0..15, 16-31 K=16..31
        FragBF bfrag[2];
#pragma unroll
        for (int n16 = 0; n16 < 2; ++n16) {
            const int n  = wn + n16 * 16 + lrow;
            const int kb = halfsel * 16;
            bfrag[n16].h[0] = *(const v8bf*)(cB + n * 32 + kb);
            bfrag[n16].h[1] = *(const v8bf*)(cB + n * 32 + kb + 8);
        }
        // ---- A fragments + WMMA ---------------------------------------
#pragma unroll
        for (int m16 = 0; m16 < 4; ++m16) {
            const int ar = wm + m16 * 16 + lrow;
            const int ak = halfsel * 16;
            FragBF afrag;
            afrag.h[0] = *(const v8bf*)(cA + ar * 32 + ak);        // K 0..7 / 16..23
            afrag.h[1] = *(const v8bf*)(cA + ar * 32 + 8 + ak);    // K 8..15 / 24..31
#pragma unroll
            for (int n16 = 0; n16 < 2; ++n16) {
                acc[m16][n16] = __builtin_amdgcn_wmma_f32_16x16x32_bf16(
                    false, afrag.f, false, bfrag[n16].f,
                    (short)0, acc[m16][n16], false, false);
            }
        }
        __syncthreads();   // all reads of buf[cur] done before it is restaged
    }

    // ---- epilogue: bias + relu + store -------------------------------
#pragma unroll
    for (int m16 = 0; m16 < 4; ++m16) {
#pragma unroll
        for (int n16 = 0; n16 < 2; ++n16) {
            const int ncol = col0 + wn + n16 * 16 + lrow;
#pragma unroll
            for (int rr = 0; rr < 8; ++rr) {
                const int mrow = row0 + wm + m16 * 16 + rr + halfsel * 8;
                float x = acc[m16][n16][rr];
                if (BIAS_MODE == 1)
                    x += bf2f(H0b[(size_t)mrow * N + ncol]);
                if (BIAS_MODE == 2)
                    x += bias_vec[ncol];
                x = fmaxf(x, 0.f);
                if (OUT_F32)
                    ((float*)Out)[(size_t)mrow * N + ncol] = x;
                else
                    ((unsigned short*)Out)[(size_t)mrow * N + ncol] = f2bf(x);
            }
        }
    }
}

// ---------------------------------------------------------------------------
// Segment sum: Macc[edge_dst[e], :] += H[e, :]   (bf16 in, f32 atomics)
// ---------------------------------------------------------------------------
__global__ __launch_bounds__(256)
void segsum_kernel(const unsigned short* __restrict__ H,
                   const int* __restrict__ edge_dst,
                   float* __restrict__ Macc) {
    const int e = blockIdx.x;
    const int a = edge_dst[e];
    const unsigned short* src = H + (size_t)e * DH;
    float* dst = Macc + (size_t)a * DH;
    const int c = threadIdx.x * 8;
    uint4 h8 = *(const uint4*)(src + c);
    const unsigned short* hs = (const unsigned short*)&h8;
#pragma unroll
    for (int i = 0; i < 8; ++i)
        atomicAdd(dst + c + i, bf2f(hs[i]));
}

// ---------------------------------------------------------------------------
// Edge message: M[e,:] = bf16( Macc[edge_src[e],:] - H[rev[e],:] )
// ---------------------------------------------------------------------------
__global__ __launch_bounds__(256)
void edge_msg_kernel(const float* __restrict__ Macc,
                     const unsigned short* __restrict__ H,
                     const int* __restrict__ edge_src,
                     const int* __restrict__ rev,
                     unsigned short* __restrict__ Mout) {
    const int e = blockIdx.x;
    const int s = edge_src[e];
    const int r = rev[e];
    const float* ma = Macc + (size_t)s * DH;
    const unsigned short* hr = H + (size_t)r * DH;
    unsigned short* out = Mout + (size_t)e * DH;
    const int c = threadIdx.x * 8;
    uint4 h8 = *(const uint4*)(hr + c);
    const unsigned short* hs = (const unsigned short*)&h8;
    float4 m0 = *(const float4*)(ma + c);
    float4 m1 = *(const float4*)(ma + c + 4);
    unsigned short o[8];
    o[0] = f2bf(m0.x - bf2f(hs[0]));
    o[1] = f2bf(m0.y - bf2f(hs[1]));
    o[2] = f2bf(m0.z - bf2f(hs[2]));
    o[3] = f2bf(m0.w - bf2f(hs[3]));
    o[4] = f2bf(m1.x - bf2f(hs[4]));
    o[5] = f2bf(m1.y - bf2f(hs[5]));
    o[6] = f2bf(m1.z - bf2f(hs[6]));
    o[7] = f2bf(m1.w - bf2f(hs[7]));
    *(uint4*)(out + c) = *(const uint4*)o;
}

// ---------------------------------------------------------------------------
// Molecule mean aggregation
// ---------------------------------------------------------------------------
__global__ __launch_bounds__(256)
void mol_acc_kernel(const float* __restrict__ Hv,
                    const int* __restrict__ batch,
                    float* __restrict__ sums,
                    float* __restrict__ counts) {
    const int a = blockIdx.x;
    const int m = batch[a];
    if (threadIdx.x == 0) atomicAdd(counts + m, 1.0f);
    const float* src = Hv + (size_t)a * DH;
    float* dst = sums + (size_t)m * DH;
    const int c = threadIdx.x * 8;
    float4 v0 = *(const float4*)(src + c);
    float4 v1 = *(const float4*)(src + c + 4);
    atomicAdd(dst + c + 0, v0.x);
    atomicAdd(dst + c + 1, v0.y);
    atomicAdd(dst + c + 2, v0.z);
    atomicAdd(dst + c + 3, v0.w);
    atomicAdd(dst + c + 4, v1.x);
    atomicAdd(dst + c + 5, v1.y);
    atomicAdd(dst + c + 6, v1.z);
    atomicAdd(dst + c + 7, v1.w);
}

__global__ __launch_bounds__(256)
void mol_div_kernel(const float* __restrict__ sums,
                    const float* __restrict__ counts,
                    float* __restrict__ out) {
    const int i = blockIdx.x * blockDim.x + threadIdx.x;  // 512*2048
    const int m = i >> 11;
    out[i] = sums[i] / fmaxf(counts[m], 1.0f);
}

// ---------------------------------------------------------------------------
// Host side
// ---------------------------------------------------------------------------
extern "C" void kernel_launch(void* const* d_in, const int* in_sizes, int n_in,
                              void* d_out, int out_size, void* d_ws, size_t ws_size,
                              hipStream_t stream) {
    (void)in_sizes; (void)n_in; (void)out_size; (void)ws_size;

    const float* V    = (const float*)d_in[0];
    const float* E    = (const float*)d_in[1];
    const int*   esrc = (const int*)d_in[2];
    const int*   edst = (const int*)d_in[3];
    const int*   rev  = (const int*)d_in[4];
    const int*   bidx = (const int*)d_in[5];
    const float* Wi   = (const float*)d_in[6];
    const float* Wh   = (const float*)d_in[7];
    const float* Wo   = (const float*)d_in[8];
    const float* bo   = (const float*)d_in[9];

    // workspace layout (bytes)
    char* ws = (char*)d_ws;
    const size_t HB = (size_t)NEDGES * DH * 2;       // 128 MiB bf16 edge matrix
    unsigned short* H0  = (unsigned short*)(ws);
    unsigned short* H   = (unsigned short*)(ws + HB);
    unsigned short* Mb  = (unsigned short*)(ws + 2 * HB);
    float*          Hv  = (float*)(ws + 2 * HB);     // alias Mb (final stage)
    float*          Macc = (float*)(ws + 3 * HB);    // f32 [NATOMS, DH] = 128 MiB
    char* p = ws + 4 * HB;
    unsigned short* Wib = (unsigned short*)p;  p += (size_t)DH * K_INIT * 2;   // [N,K] transposed
    unsigned short* Whb = (unsigned short*)p;  p += (size_t)DH * DH * 2;
    unsigned short* Wob = (unsigned short*)p;  p += (size_t)DH * K_FINAL * 2;
    float* msums  = (float*)p;                 p += (size_t)NMOLS * DH * 4;
    float* mcnt   = (float*)p;

    // 1) pack weights to transposed bf16 [N, Kpad]
    {
        long long t1 = (long long)DH * K_INIT;
        pack_wt_kernel<<<(unsigned)((t1 + 255) / 256), 256, 0, stream>>>(Wi, Wib, DV + DE, K_INIT, DH);
        long long t2 = (long long)DH * DH;
        pack_wt_kernel<<<(unsigned)((t2 + 255) / 256), 256, 0, stream>>>(Wh, Whb, DH, DH, DH);
        long long t3 = (long long)DH * K_FINAL;
        pack_wt_kernel<<<(unsigned)((t3 + 255) / 256), 256, 0, stream>>>(Wo, Wob, DV + DH, K_FINAL, DH);
    }

    // 2) H0 = relu(concat(V[src], E) @ Wi)
    {
        dim3 grid(DH / 128, NEDGES / 128);
        gemm_wmma_kernel<1, 0, 0><<<grid, 256, 0, stream>>>(
            nullptr, V, E, esrc, nullptr, Wib, nullptr, nullptr, H0, DH, K_INIT);
    }
    hipMemcpyAsync(H, H0, HB, hipMemcpyDeviceToDevice, stream);

    // 3) message-passing loop (DEPTH-1 = 4)
    for (int it = 0; it < 4; ++it) {
        hipMemsetAsync(Macc, 0, (size_t)NATOMS * DH * 4, stream);
        segsum_kernel<<<NEDGES, 256, 0, stream>>>(H, edst, Macc);
        edge_msg_kernel<<<NEDGES, 256, 0, stream>>>(Macc, H, esrc, rev, Mb);
        dim3 grid(DH / 128, NEDGES / 128);
        gemm_wmma_kernel<0, 1, 0><<<grid, 256, 0, stream>>>(
            Mb, nullptr, nullptr, nullptr, nullptr, Whb, H0, nullptr, H, DH, DH);
    }

    // 4) final segment sum into Macc (= M_v)
    hipMemsetAsync(Macc, 0, (size_t)NATOMS * DH * 4, stream);
    segsum_kernel<<<NEDGES, 256, 0, stream>>>(H, edst, Macc);

    // 5) H_v = relu(concat(V, M_v) @ Wo + b_o), f32 out
    {
        dim3 grid(DH / 128, NATOMS / 128);
        gemm_wmma_kernel<2, 2, 1><<<grid, 256, 0, stream>>>(
            nullptr, V, nullptr, nullptr, Macc, Wob, nullptr, bo, Hv, DH, K_FINAL);
    }

    // 6) mean aggregation per molecule
    hipMemsetAsync(msums, 0, (size_t)NMOLS * DH * 4, stream);
    hipMemsetAsync(mcnt, 0, (size_t)NMOLS * 4, stream);
    mol_acc_kernel<<<NATOMS, 256, 0, stream>>>(Hv, bidx, msums, mcnt);
    mol_div_kernel<<<(NMOLS * DH) / 256, 256, 0, stream>>>(msums, mcnt, (float*)d_out);
}
